// Str2Str_60790967108045
// MI455X (gfx1250) — compile-verified, matrix operand order checked by reading, not run on hardware
//
#include <hip/hip_runtime.h>
#include <hip/hip_bf16.h>

// ---------------------------------------------------------------------------
// Str2Str fused implementation for gfx1250 (wave32, WMMA 16x16x32 f16)
// B=1, N=64, L=512, d_msa=64, d_pair=128, d_state=16, top_k=64, KMIN=9
// ---------------------------------------------------------------------------

typedef __attribute__((ext_vector_type(16))) _Float16 v16h;
typedef __attribute__((ext_vector_type(8)))  float    v8f;

#define L_RES 512
#define EPS_LN 1e-5f

// ---------------------------------------------------------------------------
// WMMA fragment helpers (CDNA5 16-bit A 16x32 layout, CDNA5 ISA):
//   lanes 0-15: VGPR v in 0..3 -> K = 2v,2v+1 ; v in 4..7 -> K = 16+2(v-4),+1
//   lanes 16-31: same pattern with K offset +8
// B mirrored with lane -> N.
// ---------------------------------------------------------------------------
__device__ __forceinline__ int frag_k(int h, int lane) {
    int e = h >> 1;
    int base = (e < 4) ? (2 * e) : (16 + 2 * (e - 4));
    return base + (h & 1) + ((lane >= 16) ? 8 : 0);
}

__device__ __forceinline__ v16h load_a_frag(const _Float16* a, int stride, int kb, int lane) {
    v16h f;
    int m = lane & 15;
#pragma unroll
    for (int h = 0; h < 16; ++h) f[h] = a[m * stride + kb + frag_k(h, lane)];
    return f;
}

__device__ __forceinline__ v16h load_b_frag(const _Float16* w, int wcols, int kb, int nb, int lane) {
    v16h f;
    int n = lane & 15;
#pragma unroll
    for (int h = 0; h < 16; ++h) f[h] = w[(kb + frag_k(h, lane)) * wcols + nb + n];
    return f;
}

// C/D f32 16x16: VGPR r -> M = r (+8 for lanes 16-31), N = lane&15
__device__ __forceinline__ void store_d_frag(float* out, int stride, int nb, v8f c, int lane) {
    int n = lane & 15;
    int mo = (lane >> 4) << 3;
#pragma unroll
    for (int r = 0; r < 8; ++r) out[(r + mo) * stride + nb + n] = c[r];
}

// 16xK (LDS, f16, row-major) @ KxNcols weights (f16, row-major) -> 16 x (NT*16) f32
template <int K, int NT>
__device__ __forceinline__ void gemm16(const _Float16* a, int astride,
                                       const _Float16* w, int wcols,
                                       float* out, int ostride, int lane) {
#pragma unroll
    for (int nt = 0; nt < NT; ++nt) {
        v8f c = {};
#pragma unroll
        for (int kb = 0; kb < K; kb += 32) {
            v16h af = load_a_frag(a, astride, kb, lane);
            v16h bf = load_b_frag(w, wcols, kb, nt * 16, lane);
            c = __builtin_amdgcn_wmma_f32_16x16x32_f16(false, af, false, bf,
                                                       (short)0, c, false, false);
        }
        store_d_frag(out, ostride, nt * 16, c, lane);
    }
}

// ---------------------------------------------------------------------------
// K0: weight conversion to f16 (+ zero padding for K=69->96, N=8->16)
// ---------------------------------------------------------------------------
__global__ void k_weights(const float* __restrict__ We1, const float* __restrict__ We2,
                          const float* __restrict__ Wm1, const float* __restrict__ Wm2,
                          const float* __restrict__ Wc,
                          _Float16* We1h, _Float16* W2h, _Float16* Wm1mh,
                          _Float16* Wm2h, _Float16* Wch) {
    int t = blockIdx.x * 256 + threadIdx.x;
    if (t < 128 * 32) We1h[t] = (_Float16)We1[t];
    if (t < 96 * 32) {
        int r = t / 32, c = t % 32;
        W2h[t] = (_Float16)((r < 69) ? We2[r * 32 + c] : 0.f);
    }
    if (t < 32 * 64) {
        int r = t / 64, c = t % 64;
        Wm1mh[t] = (_Float16)Wm1[(64 + r) * 64 + c];   // rows 64..95 of Wm1
    }
    if (t < 64 * 32) Wm2h[t] = (_Float16)Wm2[t];
    if (t < 32 * 16) {
        int r = t / 16, c = t % 16;
        Wch[t] = (_Float16)((c < 8) ? Wc[r * 8 + c] : 0.f);
    }
}

// ---------------------------------------------------------------------------
// K1: per-residue fused: LN(msa col), attention over N, msa1, node MLP + LN,
//     nodeA = node@Wm1[:32], nodeB = node@Wm1[32:64], ca, v.
// One block (64 threads) per residue l.
// ---------------------------------------------------------------------------
__global__ __launch_bounds__(64) void k_node(
    const float* __restrict__ msa, const float* __restrict__ state,
    const float* __restrict__ seq1hot, const float* __restrict__ xyz,
    const float* __restrict__ g_msa, const float* __restrict__ b_msa,
    const float* __restrict__ g_state, const float* __restrict__ b_state,
    const float* __restrict__ Wq, const float* __restrict__ bq,
    const float* __restrict__ Wk, const float* __restrict__ bk,
    const float* __restrict__ Wx, const float* __restrict__ bx,
    const float* __restrict__ g_node, const float* __restrict__ b_node,
    const float* __restrict__ Wm1,
    float* node_g, float* nodeA_g, float* nodeB_g, float* ca_g, float* v_g) {
    const int l = blockIdx.x, t = threadIdx.x;
    __shared__ float msaln[64][64];
    __shared__ float qv[64], logit[64], attn[64];
    __shared__ float msa1[64], stateln[16], prenode[32], nodev[32];
    __shared__ float red[2];

    { // LN of msa[n=t, l, :]
        const float* mr = msa + ((size_t)t * L_RES + l) * 64;
        float s = 0.f, s2 = 0.f;
        for (int k = 0; k < 64; ++k) { float x = mr[k]; s += x; s2 += x * x; }
        float mu = s * (1.f / 64.f);
        float inv = rsqrtf(fmaxf(s2 * (1.f / 64.f) - mu * mu, 0.f) + EPS_LN);
        for (int k = 0; k < 64; ++k)
            msaln[t][k] = (mr[k] - mu) * inv * g_msa[k] + b_msa[k];
    }
    __syncthreads();
    { // q[c=t] = (msaln[0] @ Wq + bq) / 8
        float acc = bq[t];
        for (int k = 0; k < 64; ++k) acc += msaln[0][k] * Wq[k * 64 + t];
        qv[t] = acc * 0.125f;
    }
    __syncthreads();
    { // logit[n=t] = q . (msaln[n] @ Wk + bk)
        float acc = 0.f;
        for (int c = 0; c < 64; ++c) {
            float kc = bk[c];
            for (int k = 0; k < 64; ++k) kc += msaln[t][k] * Wk[k * 64 + c];
            acc += qv[c] * kc;
        }
        logit[t] = acc;
    }
    __syncthreads();
    if (t == 0) { // softmax over n
        float mx = logit[0];
        for (int n = 1; n < 64; ++n) mx = fmaxf(mx, logit[n]);
        float s = 0.f;
        for (int n = 0; n < 64; ++n) { attn[n] = expf(logit[n] - mx); s += attn[n]; }
        float inv = 1.f / s;
        for (int n = 0; n < 64; ++n) attn[n] *= inv;
    }
    __syncthreads();
    { // msa1[c=t]
        float acc = 0.f;
        for (int n = 0; n < 64; ++n) acc += attn[n] * msaln[n][t];
        msa1[t] = acc;
    }
    if (t == 0) { // state LN stats
        float s = 0.f, s2 = 0.f;
        for (int u = 0; u < 16; ++u) { float x = state[l * 16 + u]; s += x; s2 += x * x; }
        float mu = s * (1.f / 16.f);
        red[0] = mu;
        red[1] = rsqrtf(fmaxf(s2 * (1.f / 16.f) - mu * mu, 0.f) + EPS_LN);
    }
    __syncthreads();
    if (t < 16) stateln[t] = (state[l * 16 + t] - red[0]) * red[1] * g_state[t] + b_state[t];
    __syncthreads();
    if (t < 32) { // node pre-LN: concat(msa1, seq1hot, stateln) @ Wx + bx
        float acc = bx[t];
        for (int k = 0; k < 64; ++k) acc += msa1[k] * Wx[k * 32 + t];
        for (int s = 0; s < 21; ++s) acc += seq1hot[l * 21 + s] * Wx[(64 + s) * 32 + t];
        for (int u = 0; u < 16; ++u) acc += stateln[u] * Wx[(85 + u) * 32 + t];
        prenode[t] = acc;
    }
    __syncthreads();
    if (t == 0) {
        float s = 0.f, s2 = 0.f;
        for (int c = 0; c < 32; ++c) { float x = prenode[c]; s += x; s2 += x * x; }
        float mu = s * (1.f / 32.f);
        red[0] = mu;
        red[1] = rsqrtf(fmaxf(s2 * (1.f / 32.f) - mu * mu, 0.f) + EPS_LN);
    }
    __syncthreads();
    if (t < 32) {
        nodev[t] = (prenode[t] - red[0]) * red[1] * g_node[t] + b_node[t];
        node_g[l * 32 + t] = nodev[t];
    }
    __syncthreads();
    { // nodeA/nodeB, ch = t
        float a = 0.f, b = 0.f;
        for (int c = 0; c < 32; ++c) {
            a += nodev[c] * Wm1[c * 64 + t];
            b += nodev[c] * Wm1[(32 + c) * 64 + t];
        }
        nodeA_g[l * 64 + t] = a;
        nodeB_g[l * 64 + t] = b;
    }
    if (t < 3) ca_g[l * 3 + t] = xyz[l * 9 + 3 + t];          // atom index 1 = CA
    if (t < 9) v_g[l * 9 + t] = xyz[l * 9 + t] - xyz[l * 9 + 3 + (t % 3)];
}

// ---------------------------------------------------------------------------
// K2: top-k(64) neighbor mask per row i, OR |sep|<KMIN (off-diagonal).
// ---------------------------------------------------------------------------
__global__ __launch_bounds__(64) void k_mask(const float* __restrict__ ca, float* maskf) {
    const int i = blockIdx.x, t = threadIdx.x;
    __shared__ float Dsh[L_RES];
    __shared__ unsigned char sel[L_RES];
    __shared__ float bval[64];
    __shared__ int bidx[64];
    const float cix = ca[i * 3], ciy = ca[i * 3 + 1], ciz = ca[i * 3 + 2];
    for (int j = t; j < L_RES; j += 64) {
        float dx = ca[j * 3] - cix, dy = ca[j * 3 + 1] - ciy, dz = ca[j * 3 + 2] - ciz;
        float d2 = dx * dx + dy * dy + dz * dz;
        Dsh[j] = sqrtf(fmaxf(d2, 1e-12f)) + ((j == i) ? 999.9f : 0.f);
        sel[j] = 0;
    }
    __syncthreads();
    for (int itr = 0; itr < 64; ++itr) {
        float bv = 3e38f; int bi = 0x7fffffff;
        for (int j = t; j < L_RES; j += 64)
            if (Dsh[j] < bv) { bv = Dsh[j]; bi = j; }
        bval[t] = bv; bidx[t] = bi;
        __syncthreads();
        if (t == 0) {
            float mv = bval[0]; int mi = bidx[0];
            for (int n = 1; n < 64; ++n)
                if (bval[n] < mv || (bval[n] == mv && bidx[n] < mi)) { mv = bval[n]; mi = bidx[n]; }
            sel[mi] = 1;
            Dsh[mi] = 3e38f;
        }
        __syncthreads();
    }
    for (int j = t; j < L_RES; j += 64) {
        int sep = j - i;
        int as = sep < 0 ? -sep : sep;
        bool nb = (as < 9) && (j != i);
        maskf[(size_t)i * L_RES + j] = (sel[j] || nb) ? 1.f : 0.f;
    }
}

__global__ void k_zero(float* p, int n) {
    int i = blockIdx.x * 256 + threadIdx.x;
    if (i < n) p[i] = 0.f;
}

// ---------------------------------------------------------------------------
// K3: mega-fused edge pipeline. Grid = (L/16, L/16) tiles; block = 256 (8 waves).
// Each wave processes groups g = wave, wave+8: one i row x 16 j's = a 16-row
// WMMA tile. All GEMMs via v_wmma_f32_16x16x32_f16, intermediates in LDS.
// pair (134 MB) is the only large HBM stream: each contiguous 8 KB tile is
// loaded ONCE, coalesced float4 across all 32 lanes, into per-wave LDS
// staging; the next group's tile is prefetched (global_prefetch_b8) so its
// fetch overlaps the current WMMA chain. The f32 staging region is dead after
// the A1 conversion, so it is overlaid with sF/sM/sCoef to keep LDS ~103 KB.
// ---------------------------------------------------------------------------
__global__ __launch_bounds__(256) void k_edge(
    const float* __restrict__ pair,
    const float* __restrict__ g_pair, const float* __restrict__ b_pair,
    const float* __restrict__ be1, const float* __restrict__ g_e1, const float* __restrict__ b_e1,
    const float* __restrict__ be2, const float* __restrict__ g_e2, const float* __restrict__ b_e2,
    const float* __restrict__ Wm1, const float* __restrict__ bm1,
    const float* __restrict__ bm2, const float* __restrict__ bc,
    const float* __restrict__ nodeA, const float* __restrict__ nodeB,
    const float* __restrict__ ca, const float* __restrict__ vg,
    const float* __restrict__ maskf,
    const _Float16* __restrict__ We1h, const _Float16* __restrict__ W2h,
    const _Float16* __restrict__ Wm1mh, const _Float16* __restrict__ Wm2h,
    const _Float16* __restrict__ Wch,
    float* aggm_g, float* aggv_g) {
    __shared__ _Float16 __align__(16) sA[8][16 * 128];   // per-wave A staging (max K=128)
    __shared__ float __align__(16) sScr[8][2048];        // overlaid: sP | sF/sM/sCoef
    __shared__ float sStat[8][32][2];
    __shared__ float sDist[8][16];
    __shared__ float sDvec[8][16 * 3];
    __shared__ float aggm[16 * 32];        // per-block j-local accumulators
    __shared__ float aggv[16 * 6];

    const int wave = threadIdx.x >> 5, lane = threadIdx.x & 31;
    for (int idx = threadIdx.x; idx < 16 * 32; idx += 256) aggm[idx] = 0.f;
    for (int idx = threadIdx.x; idx < 16 * 6; idx += 256) aggv[idx] = 0.f;
    __syncthreads();

    const int it = blockIdx.y, jt = blockIdx.x;
    const int j0 = jt * 16;

    float* const sP  = sScr[wave];          // [0,2048): stage-1 pair staging
    float* const sF  = sScr[wave];          // [0,1024): GEMM f32 outputs
    float* const sM  = sScr[wave] + 1024;   // [1024,1536): m values
    float* const sCf = sScr[wave] + 1536;   // [1536,1664): coef

    for (int g = wave; g < 16; g += 8) {   // uniform: 2 iterations per wave
        const int i = it * 16 + g;
        // ---- stage 1a: coalesced cooperative load of the contiguous 8 KB tile
        {
            const float4* gsrc = (const float4*)(pair + ((size_t)i * L_RES + j0) * 128);
            float4* dst4 = (float4*)sP;
#pragma unroll
            for (int itn = 0; itn < 16; ++itn)
                dst4[itn * 32 + lane] = gsrc[itn * 32 + lane];
            if (g + 8 < 16) {  // prefetch next group's tile (one cacheline/lane)
                const char* nx = (const char*)(pair + ((size_t)(i + 8) * L_RES + j0) * 128);
                __builtin_prefetch(nx + lane * 256, 0, 3);
            }
        }
        if (lane < 16) {       // geometry per row r=lane
            const int j = j0 + lane;
            float dv0 = ca[j * 3 + 0] - ca[i * 3 + 0];
            float dv1 = ca[j * 3 + 1] - ca[i * 3 + 1];
            float dv2 = ca[j * 3 + 2] - ca[i * 3 + 2];
            sDvec[wave][lane * 3 + 0] = dv0;
            sDvec[wave][lane * 3 + 1] = dv1;
            sDvec[wave][lane * 3 + 2] = dv2;
            sDist[wave][lane] = sqrtf(dv0 * dv0 + dv1 * dv1 + dv2 * dv2);
        }
        __syncthreads();
        // ---- stage 1b: LN stats, 2 lanes per row (half-row partial sums)
        {
            const float* rowp = sP + (lane >> 1) * 128 + (lane & 1) * 64;
            float ps = 0.f, ps2 = 0.f;
            for (int c = 0; c < 64; ++c) { float x = rowp[c]; ps += x; ps2 += x * x; }
            sStat[wave][lane][0] = ps;
            sStat[wave][lane][1] = ps2;
        }
        __syncthreads();
        // ---- stage 1c: normalize -> A1 (16 x 128 f16)
        {
            const int r = lane >> 1, h = lane & 1;
            float s  = sStat[wave][r * 2][0] + sStat[wave][r * 2 + 1][0];
            float s2 = sStat[wave][r * 2][1] + sStat[wave][r * 2 + 1][1];
            float mu = s * (1.f / 128.f);
            float inv = rsqrtf(fmaxf(s2 * (1.f / 128.f) - mu * mu, 0.f) + EPS_LN);
            const float* rowp = sP + r * 128 + h * 64;
            _Float16* arow = &sA[wave][r * 128 + h * 64];
            for (int c = 0; c < 64; ++c)
                arow[c] = (_Float16)((rowp[c] - mu) * inv * g_pair[h * 64 + c] + b_pair[h * 64 + c]);
        }
        __syncthreads();
        // ---- GEMM1: (16x128) @ We1 (128x32)   (overwrites sP region via sF)
        gemm16<128, 2>(sA[wave], 128, We1h, 32, sF, 32, lane);
        __syncthreads();
        // ---- e1 = LN(C1 + be1); build A2 = [e1 | rbf36 | neighbor | 0-pad] (16 x 96)
        if (lane < 16) {
            float* row = &sF[lane * 32];
            float s = 0.f, s2 = 0.f;
            for (int c = 0; c < 32; ++c) { float x = row[c] + be1[c]; row[c] = x; s += x; s2 += x * x; }
            float mu = s * (1.f / 32.f);
            float inv = rsqrtf(fmaxf(s2 * (1.f / 32.f) - mu * mu, 0.f) + EPS_LN);
            _Float16* a2 = &sA[wave][lane * 96];
            for (int c = 0; c < 32; ++c)
                a2[c] = (_Float16)((row[c] - mu) * inv * g_e1[c] + b_e1[c]);
            float dcl = fmaxf(sDist[wave][lane], 1e-6f);   // cdist clamp sqrt(1e-12)
            for (int kk = 0; kk < 36; ++kk) {
                float tt = (dcl - (20.f / 35.f) * (float)kk) * (36.f / 20.f);
                a2[32 + kk] = (_Float16)expf(-tt * tt);
            }
            int sep = (j0 + lane) - i;
            a2[68] = (_Float16)((sep >= -1 && sep <= 1) ? (float)sep : 0.f);
            for (int c = 69; c < 96; ++c) a2[c] = (_Float16)0.f;
        }
        __syncthreads();
        // ---- GEMM2: (16x96) @ We2cat (96x32)
        gemm16<96, 2>(sA[wave], 96, W2h, 32, sF, 32, lane);
        __syncthreads();
        // ---- e2 = LN(C2 + be2) -> A3 (16 x 32 f16)
        if (lane < 16) {
            float* row = &sF[lane * 32];
            float s = 0.f, s2 = 0.f;
            for (int c = 0; c < 32; ++c) { float x = row[c] + be2[c]; row[c] = x; s += x; s2 += x * x; }
            float mu = s * (1.f / 32.f);
            float inv = rsqrtf(fmaxf(s2 * (1.f / 32.f) - mu * mu, 0.f) + EPS_LN);
            _Float16* a3 = &sA[wave][lane * 32];
            for (int c = 0; c < 32; ++c)
                a3[c] = (_Float16)((row[c] - mu) * inv * g_e2[c] + b_e2[c]);
        }
        __syncthreads();
        // ---- GEMM3: (16x32) @ Wm1[64:96] (32x64)
        gemm16<32, 4>(sA[wave], 32, Wm1mh, 64, sF, 64, lane);
        __syncthreads();
        // ---- hid = relu(C3 + nodeA[i] + nodeB[j] + dist*Wm1[96] + bm1) -> A4 (16x64)
        for (int idx = lane; idx < 16 * 64; idx += 32) {
            int r = idx >> 6, ch = idx & 63;
            int j = j0 + r;
            float h = sF[idx] + nodeA[i * 64 + ch] + nodeB[j * 64 + ch] +
                      sDist[wave][r] * Wm1[96 * 64 + ch] + bm1[ch];
            sA[wave][r * 64 + ch] = (_Float16)fmaxf(h, 0.f);
        }
        __syncthreads();
        // ---- GEMM4: (16x64) @ Wm2 (64x32)
        gemm16<64, 2>(sA[wave], 64, Wm2h, 32, sF, 32, lane);
        __syncthreads();
        // ---- m = relu(C4 + bm2) -> sM (f32) and A5 (16x32 f16)
        for (int idx = lane; idx < 16 * 32; idx += 32) {
            int c = idx & 31;
            float mv = fmaxf(sF[idx] + bm2[c], 0.f);
            sM[idx] = mv;
            sA[wave][idx] = (_Float16)mv;
        }
        __syncthreads();
        // ---- GEMM5: (16x32) @ Wc-pad (32x16); cols 0..7 valid
        gemm16<32, 1>(sA[wave], 32, Wch, 16, sF, 16, lane);
        __syncthreads();
        if (lane < 16)
            for (int c = 0; c < 8; ++c)
                sCf[lane * 8 + c] = sF[lane * 16 + c] + bc[c];
        __syncthreads();
        // ---- masked aggregation into block-local LDS accumulators
        if (lane < 16) {
            int r = lane, j = j0 + r;
            if (maskf[(size_t)i * L_RES + j] != 0.f) {
                const float* vi = &vg[i * 9];
                const float* dv = &sDvec[wave][r * 3];
                const float* cf = &sCf[r * 8];
                for (int o = 0; o < 2; ++o)
                    for (int x = 0; x < 3; ++x) {
                        float vm = cf[o * 4] * dv[x] + cf[o * 4 + 1] * vi[0 + x] +
                                   cf[o * 4 + 2] * vi[3 + x] + cf[o * 4 + 3] * vi[6 + x];
                        atomicAdd(&aggv[r * 6 + o * 3 + x], vm);
                    }
            }
        }
        for (int idx = lane; idx < 16 * 32; idx += 32) {
            int r = idx >> 5;
            if (maskf[(size_t)i * L_RES + (j0 + r)] != 0.f)
                atomicAdd(&aggm[idx], sM[idx]);
        }
        __syncthreads();
    }
    // ---- flush block-local accumulators (one global atomic per (j,ch) per tile-row)
    for (int idx = threadIdx.x; idx < 16 * 32; idx += 256) {
        int r = idx >> 5, c = idx & 31;
        atomicAdd(&aggm_g[(size_t)(j0 + r) * 32 + c], aggm[idx]);
    }
    for (int idx = threadIdx.x; idx < 16 * 6; idx += 256) {
        int r = idx / 6, c = idx % 6;
        atomicAdd(&aggv_g[(size_t)(j0 + r) * 6 + c], aggv[idx]);
    }
}

// ---------------------------------------------------------------------------
// K4: epilogue: state_out = [node|agg_m] @ Wh + bh ; rotation update of xyz.
// One block (64 threads) per residue.
// ---------------------------------------------------------------------------
__global__ __launch_bounds__(64) void k_final(
    const float* __restrict__ node, const float* __restrict__ aggm,
    const float* __restrict__ aggv, const float* __restrict__ ca,
    const float* __restrict__ vg, const float* __restrict__ Wh,
    const float* __restrict__ bh, const float* __restrict__ Wv,
    float* out) {
    const int l = blockIdx.x, t = threadIdx.x;
    __shared__ float cat[64];
    cat[t] = (t < 32) ? node[l * 32 + t] : aggm[l * 32 + (t - 32)];
    __syncthreads();
    if (t < 16) {
        float acc = bh[t];
        for (int k = 0; k < 64; ++k) acc += cat[k] * Wh[k * 16 + t];
        out[4608 + l * 16 + t] = acc;   // state_out after xyz_new (tuple order)
    }
    if (t == 0) {
        float off[2][3];
        for (int o = 0; o < 2; ++o)
            for (int x = 0; x < 3; ++x)
                off[o][x] = aggv[l * 6 + o * 3 + x] +
                            Wv[o * 3 + 0] * vg[l * 9 + 0 + x] +
                            Wv[o * 3 + 1] * vg[l * 9 + 3 + x] +
                            Wv[o * 3 + 2] * vg[l * 9 + 6 + x];
        float R0 = off[1][0], R1 = off[1][1], R2 = off[1][2];
        float ang = sqrtf(R0 * R0 + R1 * R1 + R2 * R2);
        float inv = 1.f / (ang + 1e-5f);
        float rv0 = R0 * inv, rv1 = R1 * inv, rv2 = R2 * inv;
        float cs = cosf(ang), sn = sinf(ang);
        for (int a = 0; a < 3; ++a) {
            float v0 = vg[l * 9 + a * 3 + 0];
            float v1 = vg[l * 9 + a * 3 + 1];
            float v2 = vg[l * 9 + a * 3 + 2];
            float rdv = rv0 * v0 + rv1 * v1 + rv2 * v2;
            float cx = rv1 * v2 - rv2 * v1;
            float cy = rv2 * v0 - rv0 * v2;
            float cz = rv0 * v1 - rv1 * v0;
            float n0 = (v0 - rv0 * rdv) * cs + cx * sn + rv0 * rdv;
            float n1 = (v1 - rv1 * rdv) * cs + cy * sn + rv1 * rdv;
            float n2 = (v2 - rv2 * rdv) * cs + cz * sn + rv2 * rdv;
            out[l * 9 + a * 3 + 0] = n0 + ca[l * 3 + 0] + off[0][0];
            out[l * 9 + a * 3 + 1] = n1 + ca[l * 3 + 1] + off[0][1];
            out[l * 9 + a * 3 + 2] = n2 + ca[l * 3 + 2] + off[0][2];
        }
    }
}

// ---------------------------------------------------------------------------
extern "C" void kernel_launch(void* const* d_in, const int* in_sizes, int n_in,
                              void* d_out, int out_size, void* d_ws, size_t ws_size,
                              hipStream_t stream) {
    (void)in_sizes; (void)n_in; (void)out_size; (void)ws_size;
    const float* msa     = (const float*)d_in[0];
    const float* pair    = (const float*)d_in[1];
    const float* xyz     = (const float*)d_in[2];
    const float* state   = (const float*)d_in[3];
    const float* seq1hot = (const float*)d_in[4];
    // d_in[5]=idx (arange), d_in[6]=top_k (=64): folded into the kernels.
    const float* g_msa = (const float*)d_in[7],  *b_msa = (const float*)d_in[8];
    const float* g_pair = (const float*)d_in[9], *b_pair = (const float*)d_in[10];
    const float* g_state = (const float*)d_in[11], *b_state = (const float*)d_in[12];
    const float* Wq = (const float*)d_in[13], *bq = (const float*)d_in[14];
    const float* Wk = (const float*)d_in[15], *bk = (const float*)d_in[16];
    const float* Wx = (const float*)d_in[17], *bx = (const float*)d_in[18];
    const float* We1 = (const float*)d_in[19], *be1 = (const float*)d_in[20];
    const float* We2 = (const float*)d_in[21], *be2 = (const float*)d_in[22];
    const float* g_node = (const float*)d_in[23], *b_node = (const float*)d_in[24];
    const float* g_e1 = (const float*)d_in[25], *b_e1 = (const float*)d_in[26];
    const float* g_e2 = (const float*)d_in[27], *b_e2 = (const float*)d_in[28];
    const float* Wm1 = (const float*)d_in[29], *bm1 = (const float*)d_in[30];
    const float* Wm2 = (const float*)d_in[31], *bm2 = (const float*)d_in[32];
    const float* Wc = (const float*)d_in[33], *bc = (const float*)d_in[34];
    const float* Wh = (const float*)d_in[35], *bh = (const float*)d_in[36];
    const float* Wv = (const float*)d_in[37];

    float* ws = (float*)d_ws;
    float* node_g  = ws; ws += L_RES * 32;
    float* nodeA_g = ws; ws += L_RES * 64;
    float* nodeB_g = ws; ws += L_RES * 64;
    float* ca_g    = ws; ws += L_RES * 3;
    float* v_g     = ws; ws += L_RES * 9;
    float* maskf   = ws; ws += (size_t)L_RES * L_RES;
    float* aggm_g  = ws; ws += L_RES * 32;
    float* aggv_g  = ws; ws += L_RES * 6;
    _Float16* hw = (_Float16*)ws;
    _Float16* We1h  = hw; hw += 128 * 32;
    _Float16* W2h   = hw; hw += 96 * 32;
    _Float16* Wm1mh = hw; hw += 32 * 64;
    _Float16* Wm2h  = hw; hw += 64 * 32;
    _Float16* Wch   = hw; hw += 32 * 16;

    k_weights<<<16, 256, 0, stream>>>(We1, We2, Wm1, Wm2, Wc, We1h, W2h, Wm1mh, Wm2h, Wch);
    k_node<<<L_RES, 64, 0, stream>>>(msa, state, seq1hot, xyz,
                                     g_msa, b_msa, g_state, b_state,
                                     Wq, bq, Wk, bk, Wx, bx, g_node, b_node, Wm1,
                                     node_g, nodeA_g, nodeB_g, ca_g, v_g);
    k_mask<<<L_RES, 64, 0, stream>>>(ca_g, maskf);
    k_zero<<<(L_RES * 38 + 255) / 256, 256, 0, stream>>>(aggm_g, L_RES * 38);
    k_edge<<<dim3(L_RES / 16, L_RES / 16), 256, 0, stream>>>(
        pair, g_pair, b_pair, be1, g_e1, b_e1, be2, g_e2, b_e2,
        Wm1, bm1, bm2, bc, nodeA_g, nodeB_g, ca_g, v_g, maskf,
        We1h, W2h, Wm1mh, Wm2h, Wch, aggm_g, aggv_g);
    k_final<<<L_RES, 64, 0, stream>>>(node_g, aggm_g, aggv_g, ca_g, v_g,
                                      Wh, bh, Wv, (float*)d_out);
}
